// LSTM_Encoder_21234318311662
// MI455X (gfx1250) — matrix operand order repaired and checked by reading, not compiled, not run on hardware
//
#include <hip/hip_runtime.h>

// ---------------------------------------------------------------------------
// 2-layer LSTM encoder for MI455X (gfx1250, wave32, WMMA bf16).
//   B=64, T=512, D=H=512, 4H=2048, L=2
// Phase 1: per-layer time-parallel input GEMM  gates_pre = X @ W_ih^T + b
// Phase 2: PERSISTENT recurrence kernel (one launch per layer): 128 waves
//          loop over all T steps, syncing between steps with a device-side
//          grid barrier. Gate tiles (i,f,g,o) share the WMMA C/D lane
//          mapping, so the cell nonlinearity happens entirely in registers.
// Rationale: compute (~275 GFLOP bf16) and traffic (~600 MB @ 23.3 TB/s) are
// trivial for this chip; the cost driver is inter-step synchronization, so we
// replace 1024 kernel launches with in-kernel barriers.
// ---------------------------------------------------------------------------

typedef __bf16 bf16_t;
typedef __attribute__((ext_vector_type(16))) __bf16 bf16x16;
typedef __attribute__((ext_vector_type(8)))  __bf16 bf16x8;
typedef __attribute__((ext_vector_type(8)))  float  f32x8;

#define NB   64        // batch
#define NT   512       // time steps
#define ND   512       // input dim
#define NH   512       // hidden dim
#define N4H  2048      // 4*H
#define NL   2         // layers
#define REC_BLOCKS 32  // persistent recurrence: 32 blocks x 128 thr = 128 waves

__device__ __forceinline__ bf16_t f2bf(float f) {
  union { float f; unsigned u; } in; in.f = f;
  unsigned u = in.u;
  u += 0x7FFFu + ((u >> 16) & 1u);           // round-to-nearest-even
  union { unsigned short s; bf16_t b; } out; out.s = (unsigned short)(u >> 16);
  return out.b;
}

// A-operand (16x32 MxK, 16-bit): lane m<16 holds row m, K {0..7,16..23};
// lane m+16 holds row m, K {8..15,24..31}; 2 K-values per VGPR.
__device__ __forceinline__ bf16x16 load_a_tile(const bf16_t* base, int stride, int lane) {
  const int r  = lane & 15;
  const int ko = (lane < 16) ? 0 : 8;
  const bf16_t* p = base + (size_t)r * stride + ko;
  bf16x8 lo = *(const bf16x8*)(p);
  bf16x8 hi = *(const bf16x8*)(p + 16);
  bf16x16 v;
#pragma unroll
  for (int i = 0; i < 8; ++i) { v[i] = lo[i]; v[i + 8] = hi[i]; }
  return v;
}

// B-operand (32x16 KxN, 16-bit): lane n<16 holds col n, K {0..7,16..23};
// lane n+16 holds col n, K {8..15,24..31}. NT-GEMM: col n of B == row n of W.
__device__ __forceinline__ bf16x16 load_b_tile(const bf16_t* base, int stride, int lane) {
  const int c  = lane & 15;
  const int ko = (lane < 16) ? 0 : 8;
  const bf16_t* p = base + (size_t)c * stride + ko;
  bf16x8 lo = *(const bf16x8*)(p);
  bf16x8 hi = *(const bf16x8*)(p + 16);
  bf16x16 v;
#pragma unroll
  for (int i = 0; i < 8; ++i) { v[i] = lo[i]; v[i + 8] = hi[i]; }
  return v;
}

// ---------------------------------------------------------------------------
// Big NT GEMM with bias: C[M,N] = A[M,K] @ W[N,K]^T + bias[N]   (bf16 -> f32)
// One wave computes a 32x64 output tile (2x4 WMMA accumulators).
// ---------------------------------------------------------------------------
__global__ void __launch_bounds__(256)
gemm_nt_bias_kernel(const bf16_t* __restrict__ A, const bf16_t* __restrict__ W,
                    const float* __restrict__ bias, float* __restrict__ C,
                    int M, int N, int K) {
  const int lane = threadIdx.x & 31;
  const int wave = blockIdx.x * (blockDim.x >> 5) + (threadIdx.x >> 5);
  const int tilesN = N >> 6;
  const int wm = wave / tilesN;
  const int wn = wave % tilesN;
  const int m0 = wm << 5;
  const int n0 = wn << 6;
  if (m0 >= M) return;

  f32x8 acc[2][4];
#pragma unroll
  for (int i = 0; i < 2; ++i)
#pragma unroll
    for (int j = 0; j < 4; ++j)
#pragma unroll
      for (int e = 0; e < 8; ++e) acc[i][j][e] = 0.0f;

  for (int k0 = 0; k0 < K; k0 += 32) {
    bf16x16 a[2], b[4];
#pragma unroll
    for (int i = 0; i < 2; ++i)
      a[i] = load_a_tile(A + (size_t)(m0 + 16 * i) * K + k0, K, lane);
#pragma unroll
    for (int j = 0; j < 4; ++j)
      b[j] = load_b_tile(W + (size_t)(n0 + 16 * j) * K + k0, K, lane);
#pragma unroll
    for (int i = 0; i < 2; ++i)
#pragma unroll
      for (int j = 0; j < 4; ++j)
        acc[i][j] = __builtin_amdgcn_wmma_f32_16x16x32_bf16(
            false, a[i], false, b[j], (short)0, acc[i][j], false, false);
  }

  const int rlo = (lane >> 4) << 3;   // 0 or 8
  const int cc  = lane & 15;
#pragma unroll
  for (int i = 0; i < 2; ++i)
#pragma unroll
    for (int j = 0; j < 4; ++j)
#pragma unroll
      for (int v = 0; v < 8; ++v) {
        const int row = m0 + 16 * i + v + rlo;
        const int col = n0 + 16 * j + cc;
        C[(size_t)row * N + col] = acc[i][j][v] + bias[col];
      }
}

// ---------------------------------------------------------------------------
// Device-side grid barrier (generation counter). All REC_BLOCKS blocks are
// co-resident (32 blocks on an MI455X-class part), so spinning is safe.
// ---------------------------------------------------------------------------
__device__ __forceinline__ void grid_sync(unsigned* counter, int step) {
  __threadfence();          // make this step's h/c stores visible device-wide
  __syncthreads();
  if (threadIdx.x == 0) {
    atomicAdd(counter, 1u);
    const unsigned target = (unsigned)(REC_BLOCKS * (step + 1));
    while (atomicAdd(counter, 0u) < target) {
      __builtin_amdgcn_s_sleep(1);
    }
  }
  __syncthreads();
  __threadfence();          // acquire side
}

// ---------------------------------------------------------------------------
// Persistent LSTM recurrence for one layer. 128 waves; each wave owns one
// 16x16 h-tile and accumulates its four gate tiles over K=H with WMMA.
// Loops over all T timesteps with a grid barrier between steps.
// ---------------------------------------------------------------------------
__global__ void __launch_bounds__(128)
lstm_recurrence_kernel(const float* __restrict__ gates_pre,   // [B*T, 4H]
                       const bf16_t* __restrict__ Whh,        // [4H, H]
                       bf16_t* __restrict__ hbuf,             // [2, B, H] bf16 (double buffer, [0] zeroed)
                       float* __restrict__ c,                 // [B, H] (zeroed)
                       float* __restrict__ h_f32,             // [B, H] final-h scratch
                       bf16_t* __restrict__ y_bf16,           // [B*T, H] or null
                       float* __restrict__ y_f32,             // [B, T, H] or null
                       unsigned* __restrict__ barrier_ctr) {  // zeroed before launch
  const int lane = threadIdx.x & 31;
  const int wave = blockIdx.x * (blockDim.x >> 5) + (threadIdx.x >> 5);
  const int mt = wave >> 5;    // 0..3  over B=64
  const int nt = wave & 31;    // 0..31 over H=512
  const int m0 = mt << 4;
  const int n0 = nt << 4;
  const int rlo = (lane >> 4) << 3;
  const int cc  = lane & 15;
  const size_t BH = (size_t)NB * NH;

  for (int t = 0; t < NT; ++t) {
    const bf16_t* h_in  = hbuf + (size_t)(t & 1) * BH;
    bf16_t*       h_out = hbuf + (size_t)((t + 1) & 1) * BH;

    f32x8 acc[4];
#pragma unroll
    for (int g = 0; g < 4; ++g)
#pragma unroll
      for (int v = 0; v < 8; ++v) {
        const int row = m0 + v + rlo;
        const int col = g * NH + n0 + cc;
        acc[g][v] = gates_pre[((size_t)row * NT + t) * N4H + col];
      }

    for (int k0 = 0; k0 < NH; k0 += 32) {
      bf16x16 a = load_a_tile(h_in + (size_t)m0 * NH + k0, NH, lane);
#pragma unroll
      for (int g = 0; g < 4; ++g) {
        bf16x16 b = load_b_tile(Whh + (size_t)(g * NH + n0) * NH + k0, NH, lane);
        acc[g] = __builtin_amdgcn_wmma_f32_16x16x32_bf16(
            false, a, false, b, (short)0, acc[g], false, false);
      }
    }

#pragma unroll
    for (int v = 0; v < 8; ++v) {
      const int row = m0 + v + rlo;
      const int col = n0 + cc;
      const size_t idx = (size_t)row * NH + col;
      const float ig = 1.0f / (1.0f + __expf(-acc[0][v]));
      const float fg = 1.0f / (1.0f + __expf(-acc[1][v]));
      const float gg = tanhf(acc[2][v]);
      const float og = 1.0f / (1.0f + __expf(-acc[3][v]));
      const float cn = fg * c[idx] + ig * gg;     // c touched only by this wave
      c[idx] = cn;
      const float hn = og * tanhf(cn);
      h_out[idx] = f2bf(hn);
      if (t == NT - 1) h_f32[idx] = hn;
      const size_t oidx = ((size_t)row * NT + t) * NH + col;
      if (y_bf16) y_bf16[oidx] = f2bf(hn);
      if (y_f32)  y_f32[oidx]  = hn;
    }

    if (t != NT - 1) grid_sync(barrier_ctr, t);
  }
}

// ---------------------------------------------------------------------------
// Small utility kernels
// ---------------------------------------------------------------------------
__global__ void convert_f32_bf16_kernel(const float* __restrict__ in,
                                        bf16_t* __restrict__ out, int n) {
  const int i = blockIdx.x * blockDim.x + threadIdx.x;
  if (i < n) out[i] = f2bf(in[i]);
}

__global__ void bias_sum_kernel(const float* __restrict__ a,
                                const float* __restrict__ b,
                                float* __restrict__ o, int n) {
  const int i = blockIdx.x * blockDim.x + threadIdx.x;
  if (i < n) o[i] = a[i] + b[i];
}

__global__ void zero_u32_kernel(unsigned* __restrict__ p, int n) {
  const int i = blockIdx.x * blockDim.x + threadIdx.x;
  if (i < n) p[i] = 0u;
}

// Tail of d_out: h_n [L,B,H] then c_n [L,B,H]
__global__ void final_copy_kernel(const float* __restrict__ h0,
                                  const float* __restrict__ h1,
                                  const float* __restrict__ c0,
                                  const float* __restrict__ c1,
                                  float* __restrict__ out_tail) {
  const int i = blockIdx.x * blockDim.x + threadIdx.x;  // 0 .. 4*B*H-1
  if (i >= 4 * NB * NH) return;
  const int which = i >> 15;     // B*H = 32768
  const int j = i & 32767;
  float v;
  if (which == 0)      v = h0[j];
  else if (which == 1) v = h1[j];
  else if (which == 2) v = c0[j];
  else                 v = c1[j];
  out_tail[i] = v;
}

// ---------------------------------------------------------------------------
extern "C" void kernel_launch(void* const* d_in, const int* in_sizes, int n_in,
                              void* d_out, int out_size, void* d_ws, size_t ws_size,
                              hipStream_t stream) {
  const float* x    = (const float*)d_in[0];   // [B,T,D]
  const float* Wih  = (const float*)d_in[1];   // [L,4H,D]
  const float* Whh  = (const float*)d_in[2];   // [L,4H,H]
  const float* bih  = (const float*)d_in[3];   // [L,4H]
  const float* bhh  = (const float*)d_in[4];   // [L,4H]
  float* out = (float*)d_out;                  // out[B,T,H] ++ h_n ++ c_n

  const size_t BT = (size_t)NB * NT;           // 32768
  const size_t BH = (size_t)NB * NH;           // 32768

  // ---- workspace carve ----
  char* w = (char*)d_ws;
  float*    gates  = (float*)w;    w += BT * N4H * sizeof(float);     // 256 MB
  bf16_t*   xbf    = (bf16_t*)w;   w += BT * ND * sizeof(bf16_t);     // 32 MB
  bf16_t*   y0     = (bf16_t*)w;   w += BT * NH * sizeof(bf16_t);     // 32 MB
  bf16_t*   wihb   = (bf16_t*)w;   w += (size_t)NL * N4H * ND * sizeof(bf16_t);
  bf16_t*   whhb   = (bf16_t*)w;   w += (size_t)NL * N4H * NH * sizeof(bf16_t);
  float*    bsum   = (float*)w;    w += (size_t)NL * N4H * sizeof(float);
  bf16_t*   h0buf  = (bf16_t*)w;   w += 2 * BH * sizeof(bf16_t);      // double buffer
  bf16_t*   h1buf  = (bf16_t*)w;   w += 2 * BH * sizeof(bf16_t);
  float*    cbuf   = (float*)w;    w += (size_t)NL * BH * sizeof(float);
  float*    hstate = (float*)w;    w += (size_t)NL * BH * sizeof(float);
  unsigned* barctr = (unsigned*)w; w += 64;

  // ---- conversions + bias fold ----
  {
    int n = (int)(BT * ND);
    convert_f32_bf16_kernel<<<(n + 255) / 256, 256, 0, stream>>>(x, xbf, n);
  }
  {
    int n = NL * N4H * ND;
    convert_f32_bf16_kernel<<<(n + 255) / 256, 256, 0, stream>>>(Wih, wihb, n);
  }
  {
    int n = NL * N4H * NH;
    convert_f32_bf16_kernel<<<(n + 255) / 256, 256, 0, stream>>>(Whh, whhb, n);
  }
  {
    int n = NL * N4H;
    bias_sum_kernel<<<(n + 255) / 256, 256, 0, stream>>>(bih, bhh, bsum, n);
  }
  // zero h double-buffers + c + hstate + barrier counter (contiguous region)
  {
    int nwords = (int)((2 * BH * 2 + 2 * BH * 2 + NL * BH * 4 + NL * BH * 4 + 64) / 4);
    zero_u32_kernel<<<(nwords + 255) / 256, 256, 0, stream>>>((unsigned*)h0buf, nwords);
  }

  const int gemmWaves  = (int)((BT / 32) * (N4H / 64));   // 32768 waves
  const int gemmBlocks = gemmWaves / 8;                   // 256 thr = 8 waves

  // ---- layer 0 ----
  gemm_nt_bias_kernel<<<gemmBlocks, 256, 0, stream>>>(
      xbf, wihb, bsum, gates, (int)BT, N4H, ND);
  lstm_recurrence_kernel<<<REC_BLOCKS, 128, 0, stream>>>(
      gates, whhb, h0buf, cbuf, hstate, y0, nullptr, barctr);

  // ---- layer 1 ----
  gemm_nt_bias_kernel<<<gemmBlocks, 256, 0, stream>>>(
      y0, wihb + (size_t)N4H * ND, bsum + N4H, gates, (int)BT, N4H, NH);
  zero_u32_kernel<<<1, 64, 0, stream>>>(barctr, 16);   // reset barrier counter
  lstm_recurrence_kernel<<<REC_BLOCKS, 128, 0, stream>>>(
      gates, whhb + (size_t)N4H * NH, h1buf, cbuf + BH, hstate + BH,
      nullptr, out, barctr);

  // ---- final states ----
  {
    int n = 4 * (int)BH;
    final_copy_kernel<<<(n + 255) / 256, 256, 0, stream>>>(
        hstate, hstate + BH, cbuf, cbuf + BH, out + (size_t)NB * NT * NH);
  }
}
